// MambaSSM_29987461661129
// MI455X (gfx1250) — matrix operand order repaired and checked by reading, not compiled, not run on hardware
//
#include <hip/hip_runtime.h>
#include <hip/hip_bf16.h>
#include <math.h>

// ---------------------------------------------------------------- constants
#define D_MODEL 1024
#define D_STATE 16
#define D_CONV  4
#define D_INNER 2048
#define DT_RANK 64
#define SEQ     4096
#define BATCH   4
#define MTOT    (BATCH * SEQ)   // 16384 rows

typedef __attribute__((ext_vector_type(16))) __bf16 v16bf;
typedef __attribute__((ext_vector_type(8)))  __bf16 v8bf;
typedef __attribute__((ext_vector_type(8)))  float  v8f;

// ---------------------------------------------------------------- async->LDS
#if __has_builtin(__builtin_amdgcn_global_load_async_to_lds_b128) && \
    __has_builtin(__builtin_amdgcn_global_store_async_from_lds_b128) && \
    __has_builtin(__builtin_amdgcn_s_wait_asynccnt)
#define HAVE_ASYNC_LDS 1
#else
#define HAVE_ASYNC_LDS 0
#endif

typedef int v4i __attribute__((vector_size(16)));
typedef __attribute__((address_space(1))) v4i* gp4i_t;   // global v4i*
typedef __attribute__((address_space(3))) v4i* lp4i_t;   // LDS v4i*

// generic->LDS offset: low 32 bits of a generic LDS address are the LDS byte
// offset (ISA: flat aperture maps LDS by truncating to addr[31:0]).
__device__ __forceinline__ void async_ld16(const void* g, void* l) {
#if HAVE_ASYNC_LDS
    __builtin_amdgcn_global_load_async_to_lds_b128(
        (gp4i_t)(uintptr_t)g, (lp4i_t)(uint32_t)(uintptr_t)l, 0, 0);
#else
    *(v8bf*)l = *(const v8bf*)g;
#endif
}
__device__ __forceinline__ void async_st16(void* g, const void* l) {
#if HAVE_ASYNC_LDS
    __builtin_amdgcn_global_store_async_from_lds_b128(
        (gp4i_t)(uintptr_t)g, (lp4i_t)(uint32_t)(uintptr_t)l, 0, 0);
#else
    *(v8bf*)g = *(const v8bf*)l;
#endif
}
__device__ __forceinline__ void async_wait0() {
#if HAVE_ASYNC_LDS
    __builtin_amdgcn_s_wait_asynccnt(0);
#endif
}

// ---------------------------------------------------------------- f32 -> bf16
__global__ void k_f32_to_bf16(const float* __restrict__ in,
                              __bf16* __restrict__ out, int n) {
    int i = (blockIdx.x * blockDim.x + threadIdx.x) * 8;
    if (i >= n) return;
    const v8f vf = *(const v8f*)(in + i);
    v8bf o;
#pragma unroll
    for (int r = 0; r < 8; ++r) o[r] = (__bf16)vf[r];
    *(v8bf*)(out + i) = o;
}

// ---------------------------------------------------------------- WMMA GEMM
// C[M,N](bf16) = A[M,K](bf16, lda) @ W[N,K](bf16, ldb)^T
// Per-wave tile 32(M) x 64(N): 2 A-frags x 4 B-frags -> 8 WMMA per k-step.
// Block = 8 waves as 4(M) x 2(N) -> 128 x 128 block tile.
// MODE 0: plain store. MODE 1: softplus(acc + bias[n]) (delta path).
__device__ __forceinline__ v16bf load_a_frag(const __bf16* A, int lda,
                                             int row, int k0, int lane) {
    const int half = lane >> 4, m = lane & 15;
    const __bf16* p = A + (size_t)(row + m) * lda + k0 + half * 16;
    v8bf lo = *(const v8bf*)p;          // K = k0 + half*16 .. +7
    v8bf hi = *(const v8bf*)(p + 8);    // K = k0 + half*16 + 8 .. +15
    v16bf r;
#pragma unroll
    for (int i = 0; i < 8; ++i) { r[i] = lo[i]; r[i + 8] = hi[i]; }
    return r;
}

__device__ __forceinline__ v16bf load_b_frag(const __bf16* W, int ldb,
                                             int col, int k0, int lane, int N) {
    int n = col + (lane & 15);
    if (n >= N) n = N - 1;              // clamp: garbage cols never stored
    const __bf16* p = W + (size_t)n * ldb + k0 + (lane >> 4) * 16;
    return *(const v16bf*)p;            // 16 contiguous K values per lane
}

template <int MODE>
__device__ __forceinline__ void store_tile(__bf16* C, int ldc, int row, int col,
                                           v8f acc, int lane, int N,
                                           const __bf16* bias) {
    const int half = lane >> 4;
    const int n = col + (lane & 15);
    if (n >= N) return;
    float bv = 0.f;
    if (MODE == 1) bv = (float)bias[n];
#pragma unroll
    for (int r = 0; r < 8; ++r) {
        float v = acc[r];
        if (MODE == 1) {
            v += bv;
            v = (v > 20.f) ? v : log1pf(__expf(v));   // softplus
        }
        C[(size_t)(row + r + 8 * half) * ldc + n] = (__bf16)v;
    }
}

template <int MODE>
__global__ void k_gemm_wmma(const __bf16* __restrict__ A, int lda,
                            const __bf16* __restrict__ W, int ldb,
                            __bf16* __restrict__ C, int ldc,
                            int M, int N, int K,
                            const __bf16* __restrict__ bias) {
    const int lane = threadIdx.x & 31;
    const int wave = threadIdx.x >> 5;                 // 0..7
    const int mrow = blockIdx.y * 128 + (wave >> 1) * 32;
    const int ncol = blockIdx.x * 128 + (wave & 1) * 64;
    (void)M;

    v8f acc[2][4];
#pragma unroll
    for (int i = 0; i < 2; ++i)
#pragma unroll
        for (int j = 0; j < 4; ++j) acc[i][j] = (v8f){};

    for (int k0 = 0; k0 < K; k0 += 32) {
        v16bf a0 = load_a_frag(A, lda, mrow,      k0, lane);
        v16bf a1 = load_a_frag(A, lda, mrow + 16, k0, lane);
        v16bf b[4];
#pragma unroll
        for (int j = 0; j < 4; ++j)
            b[j] = load_b_frag(W, ldb, ncol + 16 * j, k0, lane, N);
#pragma unroll
        for (int j = 0; j < 4; ++j) {
            acc[0][j] = __builtin_amdgcn_wmma_f32_16x16x32_bf16(
                false, a0, false, b[j], (short)0, acc[0][j], false, false);
            acc[1][j] = __builtin_amdgcn_wmma_f32_16x16x32_bf16(
                false, a1, false, b[j], (short)0, acc[1][j], false, false);
        }
    }
#pragma unroll
    for (int i = 0; i < 2; ++i)
#pragma unroll
        for (int j = 0; j < 4; ++j)
            store_tile<MODE>(C, ldc, mrow + 16 * i, ncol + 16 * j,
                             acc[i][j], lane, N, bias);
}

// ---------------------------------------------------------------- conv + SiLU
__global__ void k_conv_silu(const __bf16* __restrict__ xz,
                            const __bf16* __restrict__ w,      // [2048, 4]
                            const __bf16* __restrict__ bias,   // [2048]
                            __bf16* __restrict__ xconv) {      // [MTOT, 2048]
    size_t i = (size_t)blockIdx.x * blockDim.x + threadIdx.x;
    if (i >= (size_t)MTOT * D_INNER) return;
    const int c = (int)(i % D_INNER);
    const size_t bt = i / D_INNER;             // global row b*SEQ + t
    const int t = (int)(bt % SEQ);
    const size_t brow = bt - t;                // b*SEQ
    float acc = (float)bias[c];
#pragma unroll
    for (int j = 0; j < D_CONV; ++j) {
        int tt = t - (D_CONV - 1) + j;
        if (tt >= 0)
            acc += (float)w[c * D_CONV + j] *
                   (float)xz[(brow + tt) * (size_t)(2 * D_INNER) + c];
    }
    float s = acc / (1.f + __expf(-acc));      // SiLU
    xconv[i] = (__bf16)s;
}

// ---------------------------------------------------------------- SSM scan
// State-parallel: 4 lanes per channel, 4 states per lane -> 1024 waves.
// Tiles staged bf16 in LDS via async global->LDS copies (ASYNCcnt), y written
// back with async LDS->global stores; convert-on-read in the inner loop.
#define TS  32    // time-tile
#define CPB 64    // channels per block

__global__ void k_scan(const __bf16* __restrict__ xz,      // [MTOT,4096] (z at +2048)
                       const __bf16* __restrict__ xconv,   // [MTOT,2048]
                       const __bf16* __restrict__ proj,    // [MTOT,96]
                       const __bf16* __restrict__ delta,   // [MTOT,2048]
                       const __bf16* __restrict__ A_log,   // [2048,16]
                       const __bf16* __restrict__ Dskip,   // [2048]
                       __bf16* __restrict__ yout) {        // [MTOT,2048]
    const int tid  = threadIdx.x;
    const int g    = tid & 3;            // state group 0..3 (states 4g..4g+3)
    const int cloc = tid >> 2;           // 0..63
    const int c    = blockIdx.x * CPB + cloc;
    const int b    = blockIdx.y;

    __shared__ __bf16 sdt[TS][CPB];
    __shared__ __bf16 su [TS][CPB];
    __shared__ __bf16 sz [TS][CPB];
    __shared__ __bf16 sBC[TS][2 * D_STATE];   // B | C packed (proj cols 64..95)
    __shared__ __bf16 sY [TS][CPB];

    float A[4], h[4];
#pragma unroll
    for (int r = 0; r < 4; ++r) {
        A[r] = -__expf((float)A_log[(size_t)c * D_STATE + g * 4 + r]);
        h[r] = 0.f;
    }
    const float dsk = (float)Dskip[c];
    const size_t row0 = (size_t)b * SEQ;
    const int c0 = blockIdx.x * CPB;

    // copy-phase mappings (16B segments)
    const int ctt = tid >> 3, cseg = tid & 7;       // 32 rows x 8 segs (dt/u/z/y)
    const int btt = tid >> 2, bseg = tid & 3;       // 32 rows x 4 segs (B|C)

    for (int t0 = 0; t0 < SEQ; t0 += TS) {
        __syncthreads();   // previous tile fully consumed
        {
            const size_t rc = row0 + t0 + ctt;
            async_ld16(&delta[rc * D_INNER + c0 + cseg * 8], &sdt[ctt][cseg * 8]);
            async_ld16(&xconv[rc * D_INNER + c0 + cseg * 8], &su [ctt][cseg * 8]);
            async_ld16(&xz[rc * (size_t)(2 * D_INNER) + D_INNER + c0 + cseg * 8],
                       &sz[ctt][cseg * 8]);
        }
        if (tid < 128) {
            const size_t rc = row0 + t0 + btt;
            async_ld16(&proj[rc * 96 + DT_RANK + bseg * 8], &sBC[btt][bseg * 8]);
        }
        async_wait0();     // loads done; also fences prior tile's sY stores
        __syncthreads();

        for (int tt = 0; tt < TS; ++tt) {
            const float dt = (float)sdt[tt][cloc];
            const float u  = (float)su[tt][cloc];
            const float du = dt * u;
            float p = 0.f;
#pragma unroll
            for (int r = 0; r < 4; ++r) {
                float dA = __expf(dt * A[r]);
                h[r] = dA * h[r] + du * (float)sBC[tt][g * 4 + r];
                p += h[r] * (float)sBC[tt][D_STATE + g * 4 + r];
            }
            // reduce partial y over the 4 lanes of this channel
            p += __shfl_xor(p, 1, 32);
            p += __shfl_xor(p, 2, 32);
            if (g == 0) {
                float y  = p + u * dsk;                 // D skip
                float zt = (float)sz[tt][cloc];
                float gz = zt / (1.f + __expf(-zt));    // silu(z)
                sY[tt][cloc] = (__bf16)(y * gz);
            }
        }
        __syncthreads();
        async_st16(&yout[(row0 + t0 + ctt) * D_INNER + c0 + cseg * 8],
                   &sY[ctt][cseg * 8]);
        // next iteration's async_wait0 (or implicit wait at s_endpgm) covers
        // these stores before sY is rewritten.
    }
}

// ---------------------------------------------------------------- launch
extern "C" void kernel_launch(void* const* d_in, const int* in_sizes, int n_in,
                              void* d_out, int out_size, void* d_ws, size_t ws_size,
                              hipStream_t stream) {
    (void)in_sizes; (void)n_in; (void)out_size; (void)ws_size;
    const float*  x          = (const float*) d_in[0];
    const __bf16* in_proj_w  = (const __bf16*)d_in[1];   // [4096, 1024]
    const __bf16* out_proj_w = (const __bf16*)d_in[2];   // [1024, 2048]
    const __bf16* conv_w     = (const __bf16*)d_in[3];   // [2048, 1, 4]
    const __bf16* conv_b     = (const __bf16*)d_in[4];   // [2048]
    const __bf16* dt_proj_w  = (const __bf16*)d_in[5];   // [2048, 64]
    const __bf16* dt_proj_b  = (const __bf16*)d_in[6];   // [2048]
    const __bf16* x_proj_w   = (const __bf16*)d_in[7];   // [96, 2048]
    const __bf16* A_log      = (const __bf16*)d_in[8];   // [2048, 16]
    const __bf16* D_skip     = (const __bf16*)d_in[9];   // [2048]
    __bf16* out = (__bf16*)d_out;                        // [MTOT, 1024]

    __bf16* ws     = (__bf16*)d_ws;
    const size_t M = MTOT;
    __bf16* x_bf   = ws;                        // M*1024
    __bf16* xzbuf  = x_bf   + M * D_MODEL;      // M*4096
    __bf16* xconv  = xzbuf  + M * 2 * D_INNER;  // M*2048
    __bf16* projb  = xconv  + M * D_INNER;      // M*96
    __bf16* deltab = projb  + M * 96;           // M*2048
    __bf16* ybuf   = deltab + M * D_INNER;      // M*2048

    // 1. x (f32) -> bf16 (vectorized, 8 elems/thread)
    {
        int n = (int)(M * D_MODEL);
        k_f32_to_bf16<<<(n / 8 + 255) / 256, 256, 0, stream>>>(x, x_bf, n);
    }
    // 2. xz = x_bf @ in_proj_w^T        [M,1024]x[4096,1024] -> [M,4096]
    {
        dim3 g(2 * D_INNER / 128, M / 128);
        k_gemm_wmma<0><<<g, 256, 0, stream>>>(x_bf, D_MODEL, in_proj_w, D_MODEL,
                                              xzbuf, 2 * D_INNER,
                                              (int)M, 2 * D_INNER, D_MODEL, nullptr);
    }
    // 3. causal depthwise conv + SiLU
    {
        size_t n = M * D_INNER;
        k_conv_silu<<<(unsigned)((n + 255) / 256), 256, 0, stream>>>(
            xzbuf, conv_w, conv_b, xconv);
    }
    // 4. proj = x_conv @ x_proj_w^T     [M,2048]x[96,2048] -> [M,96]
    {
        dim3 g(1, M / 128);
        k_gemm_wmma<0><<<g, 256, 0, stream>>>(xconv, D_INNER, x_proj_w, D_INNER,
                                              projb, 96,
                                              (int)M, 96, D_INNER, nullptr);
    }
    // 5. delta = softplus(dt_raw @ dt_proj_w^T + b)   [M,64]x[2048,64] -> [M,2048]
    {
        dim3 g(D_INNER / 128, M / 128);
        k_gemm_wmma<1><<<g, 256, 0, stream>>>(projb, 96, dt_proj_w, DT_RANK,
                                              deltab, D_INNER,
                                              (int)M, D_INNER, DT_RANK, dt_proj_b);
    }
    // 6. SSM scan + D-skip + silu(z) gate (state-parallel, async LDS staging)
    {
        dim3 g(D_INNER / CPB, BATCH);
        k_scan<<<g, 256, 0, stream>>>(xzbuf, xconv, projb, deltab,
                                      A_log, D_skip, ybuf);
    }
    // 7. out = y @ out_proj_w^T         [M,2048]x[1024,2048] -> [M,1024]
    {
        dim3 g(D_MODEL / 128, M / 128);
        k_gemm_wmma<0><<<g, 256, 0, stream>>>(ybuf, D_INNER, out_proj_w, D_INNER,
                                              out, D_MODEL,
                                              (int)M, D_MODEL, D_INNER, nullptr);
    }
}
